// GraphEncoder_42803644072107
// MI455X (gfx1250) — matrix operand order repaired and checked by reading, not compiled, not run on hardware
//
#include <hip/hip_runtime.h>
#include <stdint.h>

#define NN 100000
#define RR 16
#define HH 64
#define LL 3
#define EE 1600000
#define MTILES (NN / 16)     // 6250 row-tiles
#define MPW 5                // m-tiles per wave (6250 = 1250 * 5)
#define WPM (MTILES / MPW)   // 1250 waves per matrix

typedef __attribute__((ext_vector_type(16))) __bf16 v16bf;
typedef __attribute__((ext_vector_type(8)))  __bf16 v8bf;
typedef __attribute__((ext_vector_type(8)))  float  v8f;

__device__ __forceinline__ __bf16 f2bf(float f) {
  union { float f; uint32_t u; } v; v.f = f;
  uint32_t r = v.u + 0x7FFFu + ((v.u >> 16) & 1u);   // round-to-nearest-even
  union { uint16_t s; __bf16 b; } o; o.s = (uint16_t)(r >> 16);
  return o.b;
}

// ---------------------------------------------------------------------------
// Pack W [L,R,64,64] + W_root [L,64,64] (row-major f32) into WMMA B-fragment
// order, bf16:  Wpack[mat][ntile(4)][kstep(2)][lane(32)][elem(16)]
// B layout (this code's contiguous k order, 16-bit B 32x16): lane<16 -> K = elem + kstep*32,
// lane>=16 -> K = 16 + elem + kstep*32; col = ntile*16 + (lane&15).
// ---------------------------------------------------------------------------
__global__ void pack_weights(const float* __restrict__ W,
                             const float* __restrict__ Wroot,
                             __bf16* __restrict__ Wpack) {
  int tid = blockIdx.x * blockDim.x + threadIdx.x;
  if (tid >= LL * (RR + 1) * 4096) return;
  int elem = tid & 15;
  int lane = (tid >> 4) & 31;
  int ks   = (tid >> 9) & 1;
  int nt   = (tid >> 10) & 3;
  int mat  = tid >> 12;
  int l = mat / (RR + 1);
  int r = mat % (RR + 1);
  int k   = ks * 32 + ((lane < 16) ? 0 : 16) + elem;
  int col = nt * 16 + (lane & 15);
  float v;
  if (r < RR) v = W[(((size_t)l * RR + r) * HH + k) * HH + col];
  else        v = Wroot[((size_t)l * HH + k) * HH + col];
  Wpack[tid] = f2bf(v);   // store offset == tid by construction
}

// x (or relu(y_prev)) -> bf16 activations, row-major [N,64]
__global__ void convert_act(const float* __restrict__ in,
                            __bf16* __restrict__ out, int relu) {
  int tid = blockIdx.x * blockDim.x + threadIdx.x;
  if (tid >= NN * HH) return;
  float v = in[tid];
  if (relu) v = fmaxf(v, 0.0f);
  out[tid] = f2bf(v);
}

__global__ void zero_f32(float* __restrict__ p, int n) {
  int tid = blockIdx.x * blockDim.x + threadIdx.x;
  if (tid < n) p[tid] = 0.0f;
}

// counts per (dst, relation) segment — edge-structure only, computed once
__global__ void count_edges(const int* __restrict__ dst,
                            const int* __restrict__ et,
                            float* __restrict__ cnt) {
  int e = blockIdx.x * blockDim.x + threadIdx.x;
  if (e >= EE) return;
  atomicAdd(&cnt[(size_t)dst[e] * RR + et[e]], 1.0f);
}

__global__ void inv_counts(float* __restrict__ cnt, int n) {
  int tid = blockIdx.x * blockDim.x + threadIdx.x;
  if (tid < n) cnt[tid] = 1.0f / fmaxf(cnt[tid], 1.0f);
}

// ---------------------------------------------------------------------------
// Grouped GEMM: out[mat] = Xbf (N x 64) @ Wmat (64 x 64), f32 accumulate.
// One wave per (mat, strip of 5 row-tiles): B fragments (8) and bias loaded
// ONCE per wave and kept in VGPRs; per tile: 2 A-frag loads, 8 WMMAs, 4 C
// stores => 40 x v_wmma_f32_16x16x32_bf16 per wave, loads overlap WMMAs.
// If isRoot: out is [N,64] and bias is added (initializes the layer output
// before the edge scatter accumulates on top).
// ---------------------------------------------------------------------------
__global__ void __launch_bounds__(256)
gemm_wmma(const __bf16* __restrict__ X, const __bf16* __restrict__ Wp,
          float* __restrict__ out, const float* __restrict__ bias,
          int numMats, int isRoot) {
  int gw   = (blockIdx.x * 256 + threadIdx.x) >> 5;
  int lane = threadIdx.x & 31;
  if (gw >= numMats * WPM) return;
  int mat   = gw / WPM;
  int strip = gw - mat * WPM;          // covers m-tiles [strip*MPW, strip*MPW+MPW)
  int lrow  = lane & 15;
  int koff  = (lane < 16) ? 0 : 16;
  int rowOff = (lane < 16) ? 0 : 8;

  // --- B fragments: loaded once, resident in VGPRs (index nt*2 + kstep) ---
  const __bf16* wm = Wp + (size_t)mat * 4096 + lane * 16;
  v16bf b[8];
#pragma unroll
  for (int i = 0; i < 8; ++i) b[i] = *(const v16bf*)(wm + i * 512);

  // --- per-lane bias for the 4 columns this lane writes ---
  float bv[4];
#pragma unroll
  for (int nt = 0; nt < 4; ++nt)
    bv[nt] = isRoot ? bias[nt * 16 + lrow] : 0.0f;

  float* od = isRoot ? out : out + (size_t)mat * NN * HH;

#pragma unroll
  for (int t = 0; t < MPW; ++t) {
    int mtile = strip * MPW + t;
    int row   = mtile * 16 + lrow;

    // A fragments (16x32 bf16, contiguous): elems 0..7 -> K=koff..,
    // elems 8..15 -> K=8+koff..
    const __bf16* xr = X + (size_t)row * HH;
    v8bf a00 = *(const v8bf*)(xr + koff);
    v8bf a01 = *(const v8bf*)(xr + 8 + koff);
    v8bf a10 = *(const v8bf*)(xr + 32 + koff);
    v8bf a11 = *(const v8bf*)(xr + 40 + koff);
    v16bf a0, a1;
#pragma unroll
    for (int i = 0; i < 8; ++i) {
      a0[i] = a00[i]; a0[8 + i] = a01[i];
      a1[i] = a10[i]; a1[8 + i] = a11[i];
    }

#pragma unroll
    for (int nt = 0; nt < 4; ++nt) {
      v8f c = {0.f, 0.f, 0.f, 0.f, 0.f, 0.f, 0.f, 0.f};
      c = __builtin_amdgcn_wmma_f32_16x16x32_bf16(false, a0, false, b[nt * 2],
                                                  (short)0, c, false, false);
      c = __builtin_amdgcn_wmma_f32_16x16x32_bf16(false, a1, false, b[nt * 2 + 1],
                                                  (short)0, c, false, false);
      int col = nt * 16 + lrow;
#pragma unroll
      for (int v = 0; v < 8; ++v) {                 // C: VGPR v -> row m0+rowOff+v
        int m = mtile * 16 + rowOff + v;
        od[(size_t)m * HH + col] = c[v] + bv[nt];
      }
    }
  }
}

// ---------------------------------------------------------------------------
// Edge scatter: y[dst] += norm[dst*R+et] * xW[et][src].  16 threads/edge,
// float4 per thread; atomics land in L2 (y is 25.6 MB << 192 MB L2).
// ---------------------------------------------------------------------------
__global__ void __launch_bounds__(256)
scatter_edges(const int* __restrict__ src, const int* __restrict__ dst,
              const int* __restrict__ et, const float* __restrict__ xW,
              const float* __restrict__ norm, float* __restrict__ y,
              int chunkStart, int chunkEnd) {
  long long t = (long long)blockIdx.x * 256 + threadIdx.x;
  int e = (int)(t >> 4);
  int part = (int)(t & 15);
  if (e >= EE) return;
  int r = et[e];
  if (r < chunkStart || r >= chunkEnd) return;
  int s = src[e], d = dst[e];
  float w = norm[(size_t)d * RR + r];
  const float4 m = *(const float4*)(xW + ((size_t)(r - chunkStart) * NN + s) * HH + part * 4);
  float* o = y + (size_t)d * HH + part * 4;
  atomicAdd(o + 0, m.x * w);
  atomicAdd(o + 1, m.y * w);
  atomicAdd(o + 2, m.z * w);
  atomicAdd(o + 3, m.w * w);
}

// ---------------------------------------------------------------------------
extern "C" void kernel_launch(void* const* d_in, const int* in_sizes, int n_in,
                              void* d_out, int out_size, void* d_ws, size_t ws_size,
                              hipStream_t stream) {
  const int*   edge_index = (const int*)d_in[0];     // [2, E]
  const int*   srcp       = edge_index;
  const int*   dstp       = edge_index + EE;
  const int*   etp        = (const int*)d_in[1];     // [E]
  const float* node_emb   = (const float*)d_in[2];   // [N, 64]
  const float* W          = (const float*)d_in[3];   // [3, 16, 64, 64]
  const float* Wroot      = (const float*)d_in[4];   // [3, 64, 64]
  const float* bias       = (const float*)d_in[5];   // [3, 64]
  float*       outp       = (float*)d_out;           // [N, 64]

  auto alignup = [](size_t x) { return (x + 255) & ~(size_t)255; };
  char* p = (char*)d_ws;
  float*  normp = (float*)p;   p += alignup((size_t)NN * RR * 4);
  __bf16* xbf   = (__bf16*)p;  p += alignup((size_t)NN * HH * 2);
  float*  ybuf  = (float*)p;   p += alignup((size_t)NN * HH * 4);
  __bf16* wpack = (__bf16*)p;  p += alignup((size_t)LL * (RR + 1) * 4096 * 2);
  float*  xw    = (float*)p;
  size_t used = (size_t)(p - (char*)d_ws);
  size_t perRel = (size_t)NN * HH * 4;
  int chunkR = 1;
  if (ws_size > used) {
    size_t c = (ws_size - used) / perRel;
    chunkR = (c < 1) ? 1 : ((c > RR) ? RR : (int)c);
  }

  auto cdiv = [](long long a, long long b) { return (unsigned)((a + b - 1) / b); };

  // --- one-time (per call) setup: packed bf16 weights + edge normalization ---
  pack_weights<<<cdiv((long long)LL * (RR + 1) * 4096, 256), 256, 0, stream>>>(W, Wroot, wpack);
  zero_f32<<<cdiv((long long)NN * RR, 256), 256, 0, stream>>>(normp, NN * RR);
  count_edges<<<cdiv(EE, 256), 256, 0, stream>>>(dstp, etp, normp);
  inv_counts<<<cdiv((long long)NN * RR, 256), 256, 0, stream>>>(normp, NN * RR);

  // --- 3 RGCN layers ---
  for (int l = 0; l < LL; ++l) {
    convert_act<<<cdiv((long long)NN * HH, 256), 256, 0, stream>>>(
        l == 0 ? node_emb : ybuf, xbf, l > 0 ? 1 : 0);

    float* ycur = (l == LL - 1) ? outp : ybuf;

    // Root transform initializes layer output: y = x @ W_root + bias
    gemm_wmma<<<cdiv((long long)WPM * 32, 256), 256, 0, stream>>>(
        xbf, wpack + (size_t)(l * (RR + 1) + RR) * 4096, ycur, bias + l * HH, 1, 1);

    for (int c0 = 0; c0 < RR; c0 += chunkR) {
      int nm = (RR - c0 < chunkR) ? (RR - c0) : chunkR;
      // xW[r] = x @ W_r for relations in this chunk (WMMA grouped GEMM)
      gemm_wmma<<<cdiv((long long)nm * WPM * 32, 256), 256, 0, stream>>>(
          xbf, wpack + (size_t)(l * (RR + 1) + c0) * 4096, xw, nullptr, nm, 0);
      // gather messages + normalized scatter-add into y
      scatter_edges<<<cdiv((long long)EE * 16, 256), 256, 0, stream>>>(
          srcp, dstp, etp, xw, normp, ycur, c0, c0 + nm);
    }
  }
}
